// CBRetriever_86449101734463
// MI455X (gfx1250) — matrix operand order repaired and checked by reading, not compiled, not run on hardware
//
#include <hip/hip_runtime.h>
#include <hip/hip_bf16.h>

// ---------------------------------------------------------------------------
// Types for CDNA5 WMMA (wave32): v_wmma_f32_16x16x32_bf16
// ---------------------------------------------------------------------------
typedef __bf16 bf16_t;
typedef __attribute__((ext_vector_type(16))) __bf16 v16bf;
typedef __attribute__((ext_vector_type(8)))  __bf16 v8bf;
typedef __attribute__((ext_vector_type(8)))  float  v8f;

#define NEG_SLOPE 0.2f

// Order-preserving float -> uint encoding so we can use integer atomicMax.
__device__ __forceinline__ unsigned f32_enc(float x) {
  unsigned u = __float_as_uint(x);
  return (u & 0x80000000u) ? ~u : (u | 0x80000000u);
}
__device__ __forceinline__ float f32_dec(unsigned u) {
  return __uint_as_float((u & 0x80000000u) ? (u ^ 0x80000000u) : ~u);
}

// ---------------------------------------------------------------------------
// Utility kernels
// ---------------------------------------------------------------------------
__global__ void k_fill_u32(unsigned* __restrict__ p, unsigned v, int n) {
  int i = blockIdx.x * blockDim.x + threadIdx.x;
  if (i < n) p[i] = v;
}

__global__ void k_f32_to_bf16(const float* __restrict__ in, bf16_t* __restrict__ out, int n) {
  int i = blockIdx.x * blockDim.x + threadIdx.x;
  if (i < n) out[i] = (bf16_t)in[i];
}

// W[K,N] f32 (row-major) -> Wt[N,K] bf16 (row-major).  Makes B-tile loads for
// WMMA fully contiguous (16B global_load_b128 per half-lane group).
__global__ void k_transpose_bf16(const float* __restrict__ W, bf16_t* __restrict__ Wt,
                                 int K, int Nn) {
  int i = blockIdx.x * blockDim.x + threadIdx.x;
  if (i >= K * Nn) return;
  int k = i / Nn, n = i - k * Nn;
  Wt[(size_t)n * K + k] = (bf16_t)W[i];
}

// ---------------------------------------------------------------------------
// GEMM: C[M,N] f32 = A[M,K] bf16 (row-major) x Wt[N,K] bf16 (i.e. W[K,N])
// One wave32 computes a 16x64 output strip: 4 f32 accumulators sharing one
// A fragment per K-step (K stepped by 32 via v_wmma_f32_16x16x32_bf16).
// -> 2 A-loads + 8 B-loads per 4 WMMAs (2.5 b128 loads / WMMA) and 4x less
//    A traffic than one-tile-per-wave.
//
// A 16x32 bf16 VGPR layout (this code's contiguous k order, matched by B): lane = half*16 + m;
//   elems 0..7  <-> K = k0 + half*16 .. +7
//   elems 8..15 <-> K = k0 + half*16 + 8 .. +7     -> two contiguous 16B loads.
// B 32x16 bf16: lane = half*16 + n; elems 0..15 <-> K = k0 + half*16 .. +15
//   (column n contiguous in Wt row)               -> two contiguous 16B loads.
// C/D 16x16 f32: lane = half*16 + n; VGPR j <-> row m = j + half*8.
// ---------------------------------------------------------------------------
__global__ void k_gemm_wmma(const bf16_t* __restrict__ A, const bf16_t* __restrict__ Wt,
                            float* __restrict__ C, int M, int K, int Nn) {
  const int lane   = threadIdx.x & 31;
  const int wave   = threadIdx.x >> 5;
  const int tilesN = Nn >> 6;                 // 64-wide N strips
  const int tiles  = (M >> 4) * tilesN;
  const int t = blockIdx.x * 8 + wave;        // 8 waves / 256-thread block
  if (t >= tiles) return;                     // wave-uniform: EXEC stays all-1s
  const int tm   = t / tilesN;
  const int tn   = (t - tm * tilesN) * 4;     // first 16-col tile of the strip
  const int half = lane >> 4;
  const int mr   = lane & 15;                 // A row / B column / C column

  const bf16_t* arow = A + (size_t)(tm * 16 + mr) * K;
  const bf16_t* brow0 = Wt + (size_t)((tn + 0) * 16 + mr) * K;
  const bf16_t* brow1 = Wt + (size_t)((tn + 1) * 16 + mr) * K;
  const bf16_t* brow2 = Wt + (size_t)((tn + 2) * 16 + mr) * K;
  const bf16_t* brow3 = Wt + (size_t)((tn + 3) * 16 + mr) * K;

  v8f acc0 = {}, acc1 = {}, acc2 = {}, acc3 = {};
  for (int k0 = 0; k0 < K; k0 += 32) {
    v8bf a0 = *(const v8bf*)(arow + k0 + half * 16);
    v8bf a1 = *(const v8bf*)(arow + k0 + half * 16 + 8);
    v16bf a = __builtin_shufflevector(a0, a1, 0,1,2,3,4,5,6,7,8,9,10,11,12,13,14,15);

    const int kb = k0 + half * 16;
    v8bf b0l = *(const v8bf*)(brow0 + kb), b0h = *(const v8bf*)(brow0 + kb + 8);
    v8bf b1l = *(const v8bf*)(brow1 + kb), b1h = *(const v8bf*)(brow1 + kb + 8);
    v8bf b2l = *(const v8bf*)(brow2 + kb), b2h = *(const v8bf*)(brow2 + kb + 8);
    v8bf b3l = *(const v8bf*)(brow3 + kb), b3h = *(const v8bf*)(brow3 + kb + 8);
    v16bf b0 = __builtin_shufflevector(b0l, b0h, 0,1,2,3,4,5,6,7,8,9,10,11,12,13,14,15);
    v16bf b1 = __builtin_shufflevector(b1l, b1h, 0,1,2,3,4,5,6,7,8,9,10,11,12,13,14,15);
    v16bf b2 = __builtin_shufflevector(b2l, b2h, 0,1,2,3,4,5,6,7,8,9,10,11,12,13,14,15);
    v16bf b3 = __builtin_shufflevector(b3l, b3h, 0,1,2,3,4,5,6,7,8,9,10,11,12,13,14,15);

    acc0 = __builtin_amdgcn_wmma_f32_16x16x32_bf16(false, a, false, b0, (short)0, acc0, false, false);
    acc1 = __builtin_amdgcn_wmma_f32_16x16x32_bf16(false, a, false, b1, (short)0, acc1, false, false);
    acc2 = __builtin_amdgcn_wmma_f32_16x16x32_bf16(false, a, false, b2, (short)0, acc2, false, false);
    acc3 = __builtin_amdgcn_wmma_f32_16x16x32_bf16(false, a, false, b3, (short)0, acc3, false, false);
  }

  float* cbase = C + (size_t)(tm * 16 + half * 8) * Nn + tn * 16 + mr;
#pragma unroll
  for (int j = 0; j < 8; ++j) {
    size_t r = (size_t)j * Nn;
    cbase[r]      = acc0[j];
    cbase[r + 16] = acc1[j];
    cbase[r + 32] = acc2[j];
    cbase[r + 48] = acc3[j];
  }
}

// ---------------------------------------------------------------------------
// Edge pass 1: e[e,h] = att[h] . leaky_relu(xl[src] + xr[tgt]);  segment max.
// ---------------------------------------------------------------------------
__global__ void k_edge_logits(const float* __restrict__ xl, const float* __restrict__ xr,
                              const float* __restrict__ att, const int* __restrict__ ei,
                              float* __restrict__ ebuf, unsigned* __restrict__ menc,
                              int Ee, int H, int Cc) {
  int idx = blockIdx.x * blockDim.x + threadIdx.x;
  if (idx >= Ee * H) return;
  int e = idx / H, h = idx - e * H;
  int src = ei[e], tgt = ei[Ee + e];
  const float4* pl = (const float4*)(xl + ((size_t)src * H + h) * Cc);
  const float4* pr = (const float4*)(xr + ((size_t)tgt * H + h) * Cc);
  const float4* pa = (const float4*)(att + (size_t)h * Cc);
  float acc = 0.f;
  for (int c = 0; c < (Cc >> 2); ++c) {
    float4 l = pl[c], r = pr[c], av = pa[c];
    float vx = l.x + r.x; vx = vx > 0.f ? vx : vx * NEG_SLOPE;
    float vy = l.y + r.y; vy = vy > 0.f ? vy : vy * NEG_SLOPE;
    float vz = l.z + r.z; vz = vz > 0.f ? vz : vz * NEG_SLOPE;
    float vw = l.w + r.w; vw = vw > 0.f ? vw : vw * NEG_SLOPE;
    acc += av.x * vx + av.y * vy + av.z * vz + av.w * vw;
  }
  ebuf[idx] = acc;
  atomicMax(&menc[(size_t)tgt * H + h], f32_enc(acc));
}

// Edge pass 2: ex = exp(e - max[tgt]);  segment sum.
__global__ void k_edge_exp(float* __restrict__ ebuf, const unsigned* __restrict__ menc,
                           float* __restrict__ sbuf, const int* __restrict__ ei,
                           int Ee, int H) {
  int idx = blockIdx.x * blockDim.x + threadIdx.x;
  if (idx >= Ee * H) return;
  int e = idx / H, h = idx - e * H;
  int tgt = ei[Ee + e];
  float ex = __expf(ebuf[idx] - f32_dec(menc[(size_t)tgt * H + h]));
  ebuf[idx] = ex;
  atomicAdd(&sbuf[(size_t)tgt * H + h], ex);
}

// Edge pass 3: out[tgt,h,c] += (ex/s[tgt,h]) * xl[src,h,c]
__global__ void k_edge_aggr(const float* __restrict__ ebuf, const float* __restrict__ sbuf,
                            const float* __restrict__ xl, const int* __restrict__ ei,
                            float* __restrict__ agg, int Ee, int H, int Cc) {
  int idx = blockIdx.x * blockDim.x + threadIdx.x;
  if (idx >= Ee * H * Cc) return;
  int c  = idx % Cc;
  int eh = idx / Cc;
  int e  = eh / H, h = eh - e * H;
  int src = ei[e], tgt = ei[Ee + e];
  float alpha = ebuf[eh] / sbuf[(size_t)tgt * H + h];
  float v = alpha * xl[((size_t)src * H + h) * Cc + c];
  atomicAdd(&agg[((size_t)tgt * H + h) * Cc + c], v);
}

// bias (+ optional ReLU) in f32, plus bf16 copy feeding next layer's WMMA.
__global__ void k_bias_act(float* __restrict__ agg, const float* __restrict__ bias,
                           bf16_t* __restrict__ xb, int n, int HC, int relu) {
  int i = blockIdx.x * blockDim.x + threadIdx.x;
  if (i >= n) return;
  float v = agg[i] + bias[i % HC];
  if (relu) v = fmaxf(v, 0.f);
  agg[i] = v;
  xb[i] = (bf16_t)v;
}

// Per-graph sum / count / max pooling.
__global__ void k_pool(const float* __restrict__ h, const int* __restrict__ batch,
                       float* __restrict__ psum, unsigned* __restrict__ pmax,
                       float* __restrict__ pcnt, int Nn, int F) {
  int i = blockIdx.x * blockDim.x + threadIdx.x;
  if (i >= Nn * F) return;
  int node = i / F, f = i - node * F;
  int g = batch[node];
  float v = h[i];
  atomicAdd(&psum[(size_t)g * F + f], v);
  atomicMax(&pmax[(size_t)g * F + f], f32_enc(v));
  if (f == 0) atomicAdd(&pcnt[g], 1.f);
}

// pooled [G, 2F] @ Wlin [2F, OUT] + blin
__global__ void k_final(const float* __restrict__ psum, const unsigned* __restrict__ pmax,
                        const float* __restrict__ pcnt, const float* __restrict__ Wlin,
                        const float* __restrict__ blin, float* __restrict__ out,
                        int Gg, int F, int OUT) {
  int tid = blockIdx.x * blockDim.x + threadIdx.x;
  if (tid >= Gg * OUT) return;
  int g = tid / OUT, o = tid - g * OUT;
  float inv = 1.f / fmaxf(pcnt[g], 1.f);
  float acc = blin[o];
  for (int j = 0; j < F; ++j)
    acc += (psum[(size_t)g * F + j] * inv) * Wlin[(size_t)j * OUT + o];
  for (int j = 0; j < F; ++j)
    acc += f32_dec(pmax[(size_t)g * F + j]) * Wlin[(size_t)(F + j) * OUT + o];
  out[(size_t)g * OUT + o] = acc;
}

// ---------------------------------------------------------------------------
// Host orchestration
// ---------------------------------------------------------------------------
extern "C" void kernel_launch(void* const* d_in, const int* in_sizes, int n_in,
                              void* d_out, int out_size, void* d_ws, size_t ws_size,
                              hipStream_t stream) {
  (void)in_sizes; (void)n_in; (void)out_size; (void)ws_size;
  const float* x     = (const float*)d_in[0];
  const int*   ei    = (const int*)d_in[1];   // [2,E] flat: src=ei[e], tgt=ei[E+e]
  const int*   batch = (const int*)d_in[3];
  const float* Wl[3]  = {(const float*)d_in[4],  (const float*)d_in[8],  (const float*)d_in[12]};
  const float* Wr[3]  = {(const float*)d_in[5],  (const float*)d_in[9],  (const float*)d_in[13]};
  const float* att[3] = {(const float*)d_in[6],  (const float*)d_in[10], (const float*)d_in[14]};
  const float* bia[3] = {(const float*)d_in[7],  (const float*)d_in[11], (const float*)d_in[15]};
  const float* Wlin  = (const float*)d_in[16];
  const float* blin  = (const float*)d_in[17];
  float* out = (float*)d_out;

  const int Nn = 20000, Ee = 320000, Gg = 64, OUT = 16, Cc = 64;
  const int Kd[3]  = {256, 256, 128};
  const int HCd[3] = {256, 128, 64};
  const int Hd[3]  = {4, 2, 1};

  // Bump allocator over workspace (total ~78 MB).
  char* ws = (char*)d_ws;
  size_t off = 0;
  auto alloc = [&](size_t bytes) -> char* {
    off = (off + 255) & ~(size_t)255;
    char* p = ws + off; off += bytes; return p;
  };
  bf16_t*   xb   = (bf16_t*)alloc((size_t)Nn * 256 * 2);   // layer input (bf16)
  bf16_t*   wlt  = (bf16_t*)alloc((size_t)256 * 256 * 2);  // transposed weights
  bf16_t*   wrt  = (bf16_t*)alloc((size_t)256 * 256 * 2);
  float*    xl   = (float*)alloc((size_t)Nn * 256 * 4);
  float*    xr   = (float*)alloc((size_t)Nn * 256 * 4);
  float*    agg  = (float*)alloc((size_t)Nn * 256 * 4);
  float*    ebuf = (float*)alloc((size_t)Ee * 4 * 4);
  unsigned* menc = (unsigned*)alloc((size_t)Nn * 4 * 4);
  float*    sbuf = (float*)alloc((size_t)Nn * 4 * 4);
  float*    psum = (float*)alloc((size_t)Gg * 64 * 4);
  unsigned* pmax = (unsigned*)alloc((size_t)Gg * 64 * 4);
  float*    pcnt = (float*)alloc((size_t)Gg * 4);

  const int TPB = 256;
  auto nb = [](long n, int tpb) { return (int)((n + tpb - 1) / tpb); };

  k_f32_to_bf16<<<nb((long)Nn * 256, TPB), TPB, 0, stream>>>(x, xb, Nn * 256);

  for (int l = 0; l < 3; ++l) {
    const int K = Kd[l], HC = HCd[l], H = Hd[l];
    k_transpose_bf16<<<nb((long)K * HC, TPB), TPB, 0, stream>>>(Wl[l], wlt, K, HC);
    k_transpose_bf16<<<nb((long)K * HC, TPB), TPB, 0, stream>>>(Wr[l], wrt, K, HC);

    const int tiles = (Nn / 16) * (HC / 64);          // 16x64 strip per wave
    const int gblk  = (tiles + 7) / 8;                // 8 waves (wave32) per block
    k_gemm_wmma<<<gblk, 256, 0, stream>>>(xb, wlt, xl, Nn, K, HC);
    k_gemm_wmma<<<gblk, 256, 0, stream>>>(xb, wrt, xr, Nn, K, HC);

    k_fill_u32<<<nb((long)Nn * H, TPB), TPB, 0, stream>>>(menc, 0u, Nn * H);
    k_fill_u32<<<nb((long)Nn * H, TPB), TPB, 0, stream>>>((unsigned*)sbuf, 0u, Nn * H);
    k_fill_u32<<<nb((long)Nn * HC, TPB), TPB, 0, stream>>>((unsigned*)agg, 0u, Nn * HC);

    k_edge_logits<<<nb((long)Ee * H, TPB), TPB, 0, stream>>>(xl, xr, att[l], ei, ebuf, menc, Ee, H, Cc);
    k_edge_exp<<<nb((long)Ee * H, TPB), TPB, 0, stream>>>(ebuf, menc, sbuf, ei, Ee, H);
    k_edge_aggr<<<nb((long)Ee * H * Cc, TPB), TPB, 0, stream>>>(ebuf, sbuf, xl, ei, agg, Ee, H, Cc);

    k_bias_act<<<nb((long)Nn * HC, TPB), TPB, 0, stream>>>(agg, bia[l], xb, Nn * HC, HC, (l < 2) ? 1 : 0);
  }

  // Per-graph mean/max pooling + final linear head.
  k_fill_u32<<<nb((long)Gg * 64, TPB), TPB, 0, stream>>>((unsigned*)psum, 0u, Gg * 64);
  k_fill_u32<<<nb((long)Gg * 64, TPB), TPB, 0, stream>>>(pmax, 0x00800000u /*enc(-FLT_MAX)*/, Gg * 64);
  k_fill_u32<<<1, 64, 0, stream>>>((unsigned*)pcnt, 0u, Gg);
  k_pool<<<nb((long)Nn * 64, TPB), TPB, 0, stream>>>(agg, batch, psum, pmax, pcnt, Nn, 64);
  k_final<<<nb((long)Gg * OUT, TPB), TPB, 0, stream>>>(psum, pmax, pcnt, Wlin, blin, out, Gg, 64, OUT);
}
